// Transformer_49572512530941
// MI455X (gfx1250) — matrix operand order repaired and checked by reading, not compiled, not run on hardware
//
#include <hip/hip_runtime.h>
#include <hip/hip_bf16.h>
#include <math.h>

// ---------------------------------------------------------------------------
// Types / WMMA helpers (gfx1250, wave32)
// ---------------------------------------------------------------------------
typedef __bf16 bf16;
typedef __bf16 v16bf __attribute__((ext_vector_type(16)));
typedef __bf16 v8bf  __attribute__((ext_vector_type(8)));
typedef __bf16 v2bf  __attribute__((ext_vector_type(2)));
typedef float  v8f   __attribute__((ext_vector_type(8)));

#define DEV __device__ __forceinline__
#define USE_ASYNC_LDS_COPY 1

static __device__ const float kEps = 1e-5f;

// Tiles stored in LDS pre-swizzled into WMMA fragment order: each lane's 16
// bf16 elements are 32 contiguous bytes at tile + lane*16. TILE = 512 elem.
#define TILE 512

// 16x32 bf16 A-layout, contiguous k order (k = 16*half + s): element (row r, col k) -> lds slot.
DEV int sw_a(int r, int k) {
  return (r + (k & 16)) * 16 + (k & 15);
}
// Inverse of the 32x16 bf16 B-layout: element (row k, col n) -> lds slot.
DEV int sw_b(int k, int n) {
  return (n + (k & 16)) * 16 + (k & 15);
}

// One fragment = 32 contiguous bytes per lane -> 2x ds_load_b128.
DEV v16bf frag_ld(const bf16* tile, int lane) {
  return *(const v16bf*)(tile + lane * 16);
}

DEV v8f wmma_bf16(v16bf a, v16bf b, v8f c) {
  return __builtin_amdgcn_wmma_f32_16x16x32_bf16(
      false, a, false, b, (short)0, c, false, false);
}

DEV v8f v8f_zero() { v8f z = {0.f,0.f,0.f,0.f,0.f,0.f,0.f,0.f}; return z; }

// ---------------------------------------------------------------------------
// Problem constants
// ---------------------------------------------------------------------------
#define BATCH 4
#define CIN   128
#define NPTS  2048
#define DIM   256
#define PH    64
#define AH    1024
#define KNN   16
#define BN    (BATCH * NPTS)   // 8192

// ---------------------------------------------------------------------------
// Kernel 1: brute-force kNN (top-16 ascending squared distance, incl. self)
// ---------------------------------------------------------------------------
__global__ __launch_bounds__(256) void knn_topk(const float* __restrict__ pos,
                                                int* __restrict__ idx) {
  __shared__ float px[256], py[256], pz[256];
  int tid = threadIdx.x;
  int b   = blockIdx.x >> 3;
  int n   = ((blockIdx.x & 7) << 8) + tid;

  const float* pb = pos + (size_t)b * 3 * NPTS;
  float qx = pb[n], qy = pb[NPTS + n], qz = pb[2 * NPTS + n];

  float bd[KNN]; int bi[KNN];
#pragma unroll
  for (int i = 0; i < KNN; ++i) { bd[i] = 3.4e38f; bi[i] = 0; }

  for (int t = 0; t < NPTS / 256; ++t) {
    __syncthreads();
    int c = t * 256 + tid;
    px[tid] = pb[c]; py[tid] = pb[NPTS + c]; pz[tid] = pb[2 * NPTS + c];
    __syncthreads();
    for (int j = 0; j < 256; ++j) {
      float dx = qx - px[j], dy = qy - py[j], dz = qz - pz[j];
      float d  = dx * dx + dy * dy + dz * dz;
      if (d < bd[KNN - 1]) {
        bd[KNN - 1] = d; bi[KNN - 1] = t * 256 + j;
#pragma unroll
        for (int k = KNN - 1; k > 0; --k) {
          if (bd[k] < bd[k - 1]) {
            float td = bd[k]; bd[k] = bd[k - 1]; bd[k - 1] = td;
            int   ti = bi[k]; bi[k] = bi[k - 1]; bi[k - 1] = ti;
          }
        }
      }
    }
  }
#pragma unroll
  for (int k = 0; k < KNN; ++k) idx[((size_t)b * NPTS + n) * KNN + k] = bi[k];
}

// ---------------------------------------------------------------------------
// Kernel 2: h = bf16(Ws(256x128) @ x + bs). 256x64 tile per block.
// ---------------------------------------------------------------------------
__global__ __launch_bounds__(256, 1) void gemm_s(const float* __restrict__ x,
                                                 const float* __restrict__ Ws,
                                                 const float* __restrict__ bs,
                                                 bf16* __restrict__ h) {
  __shared__ __attribute__((aligned(32))) bf16 Wt[16 * TILE];
  __shared__ __attribute__((aligned(32))) bf16 Xt[4 * TILE];
  int tid = threadIdx.x, wave = tid >> 5, lane = tid & 31;
  int col0 = blockIdx.x * 64;
  int b = col0 >> 11, n0 = col0 & (NPTS - 1);

  v8f acc[8];
#pragma unroll
  for (int i = 0; i < 8; ++i) acc[i] = v8f_zero();

  for (int kc = 0; kc < CIN / 32; ++kc) {
    __syncthreads();
#pragma unroll
    for (int i = 0; i < 16; ++i) {
      int e2 = i * 256 + tid, r = e2 >> 4, c = (e2 & 15) << 1;
      const float* src = Ws + (size_t)r * CIN + kc * 32 + c;
      v2bf p = {(bf16)src[0], (bf16)src[1]};
      *(v2bf*)(Wt + (r >> 4) * TILE + sw_a(r & 15, c)) = p;
    }
#pragma unroll
    for (int i = 0; i < 8; ++i) {
      int e = i * 256 + tid, kr = e >> 6, nc = e & 63;
      Xt[(nc >> 4) * TILE + sw_b(kr, nc & 15)] =
          (bf16)x[((size_t)b * CIN + kc * 32 + kr) * NPTS + n0 + nc];
    }
    __syncthreads();
#pragma unroll
    for (int t = 0; t < 2; ++t) {
      v16bf a = frag_ld(Wt + (wave * 2 + t) * TILE, lane);
#pragma unroll
      for (int ct = 0; ct < 4; ++ct)
        acc[t * 4 + ct] = wmma_bf16(a, frag_ld(Xt + ct * TILE, lane), acc[t * 4 + ct]);
    }
  }
  int mbase = (lane >> 4) << 3, nn = lane & 15;
#pragma unroll
  for (int t = 0; t < 2; ++t)
#pragma unroll
    for (int ct = 0; ct < 4; ++ct)
#pragma unroll
      for (int r = 0; r < 8; ++r) {
        int row = (wave * 2 + t) * 16 + mbase + r;
        int col = n0 + ct * 16 + nn;
        h[((size_t)b * DIM + row) * NPTS + col] = (bf16)(acc[t * 4 + ct][r] + bs[row]);
      }
}

// ---------------------------------------------------------------------------
// Kernel 3: out(f32) = W(256x256) @ h(bf16) + bias   (q, k, v)
// ---------------------------------------------------------------------------
__global__ __launch_bounds__(256, 1) void gemm_qkv(const bf16* __restrict__ h,
                                                   const float* __restrict__ W,
                                                   const float* __restrict__ bias,
                                                   float* __restrict__ out) {
  __shared__ __attribute__((aligned(32))) bf16 Wt[16 * TILE];
  __shared__ __attribute__((aligned(32))) bf16 Ht[4 * TILE];
  int tid = threadIdx.x, wave = tid >> 5, lane = tid & 31;
  int col0 = blockIdx.x * 64;
  int b = col0 >> 11, n0 = col0 & (NPTS - 1);

  v8f acc[8];
#pragma unroll
  for (int i = 0; i < 8; ++i) acc[i] = v8f_zero();

  for (int kc = 0; kc < DIM / 32; ++kc) {
    __syncthreads();
#pragma unroll
    for (int i = 0; i < 16; ++i) {
      int e2 = i * 256 + tid, r = e2 >> 4, c = (e2 & 15) << 1;
      const float* src = W + (size_t)r * DIM + kc * 32 + c;
      v2bf p = {(bf16)src[0], (bf16)src[1]};
      *(v2bf*)(Wt + (r >> 4) * TILE + sw_a(r & 15, c)) = p;
    }
#pragma unroll
    for (int i = 0; i < 8; ++i) {
      int e = i * 256 + tid, kr = e >> 6, nc = e & 63;
      Ht[(nc >> 4) * TILE + sw_b(kr, nc & 15)] =
          h[((size_t)b * DIM + kc * 32 + kr) * NPTS + n0 + nc];
    }
    __syncthreads();
#pragma unroll
    for (int t = 0; t < 2; ++t) {
      v16bf a = frag_ld(Wt + (wave * 2 + t) * TILE, lane);
#pragma unroll
      for (int ct = 0; ct < 4; ++ct)
        acc[t * 4 + ct] = wmma_bf16(a, frag_ld(Ht + ct * TILE, lane), acc[t * 4 + ct]);
    }
  }
  int mbase = (lane >> 4) << 3, nn = lane & 15;
#pragma unroll
  for (int t = 0; t < 2; ++t)
#pragma unroll
    for (int ct = 0; ct < 4; ++ct)
#pragma unroll
      for (int r = 0; r < 8; ++r) {
        int row = (wave * 2 + t) * 16 + mbase + r;
        int col = n0 + ct * 16 + nn;
        out[((size_t)b * DIM + row) * NPTS + col] = acc[t * 4 + ct][r] + bias[row];
      }
}

// ---------------------------------------------------------------------------
// Kernel 4: pe = pW2 @ relu(bn(pW1 @ pos_rel + pb1)) + pb2 -> bf16
// ---------------------------------------------------------------------------
__global__ __launch_bounds__(256, 1) void pe_mlp(
    const float* __restrict__ pos, const int* __restrict__ idx,
    const float* __restrict__ pW1, const float* __restrict__ pb1,
    const float* __restrict__ pg,  const float* __restrict__ pbe,
    const float* __restrict__ pm,  const float* __restrict__ pv,
    const float* __restrict__ pW2, const float* __restrict__ pb2,
    bf16* __restrict__ pe) {
  __shared__ float rel[3][128];
  __shared__ __attribute__((aligned(32))) bf16 Hs[16 * TILE];   // (kc,ct) B-tiles
  __shared__ __attribute__((aligned(32))) bf16 W2s[32 * TILE];  // (rt,kc) A-tiles
  int tid = threadIdx.x, wave = tid >> 5, lane = tid & 31;
  int b = blockIdx.x >> 8, p0 = (blockIdx.x & 255) * 8;

#pragma unroll
  for (int i = 0; i < 32; ++i) {
    int e2 = i * 256 + tid, r = e2 >> 5, c = (e2 & 31) << 1;
    const float* src = pW2 + (size_t)r * PH + c;
    v2bf p = {(bf16)src[0], (bf16)src[1]};
    *(v2bf*)(W2s + ((r >> 4) * 2 + (c >> 5)) * TILE + sw_a(r & 15, c & 31)) = p;
  }
  if (tid < 128) {
    int p = tid >> 4, kk = tid & 15;
    int point = p0 + p;
    int j = idx[((size_t)b * NPTS + point) * KNN + kk];
    const float* pb = pos + (size_t)b * 3 * NPTS;
#pragma unroll
    for (int c = 0; c < 3; ++c) rel[c][tid] = pb[c * NPTS + point] - pb[c * NPTS + j];
  }
  __syncthreads();
#pragma unroll
  for (int i = 0; i < 32; ++i) {
    int e = i * 256 + tid, ph = e >> 7, col = e & 127;
    float hr = pb1[ph] + pW1[ph * 3] * rel[0][col] + pW1[ph * 3 + 1] * rel[1][col]
             + pW1[ph * 3 + 2] * rel[2][col];
    float sc = pg[ph] * rsqrtf(pv[ph] + kEps);
    float hv = (hr - pm[ph]) * sc + pbe[ph];
    Hs[((ph >> 5) * 8 + (col >> 4)) * TILE + sw_b(ph & 31, col & 15)] =
        (bf16)fmaxf(hv, 0.f);
  }
  __syncthreads();

  int mbase = (lane >> 4) << 3, nn = lane & 15;
#pragma unroll
  for (int t = 0; t < 2; ++t) {
    int rt = wave * 2 + t;
    v8f acc[8];
#pragma unroll
    for (int i = 0; i < 8; ++i) acc[i] = v8f_zero();
#pragma unroll
    for (int kc = 0; kc < 2; ++kc) {
      v16bf a = frag_ld(W2s + (rt * 2 + kc) * TILE, lane);
#pragma unroll
      for (int ct = 0; ct < 8; ++ct)
        acc[ct] = wmma_bf16(a, frag_ld(Hs + (kc * 8 + ct) * TILE, lane), acc[ct]);
    }
#pragma unroll
    for (int ct = 0; ct < 8; ++ct)
#pragma unroll
      for (int r = 0; r < 8; ++r) {
        int row = rt * 16 + mbase + r;
        pe[(((size_t)b * DIM + row) * NPTS + p0 + ct) * KNN + nn] =
            (bf16)(acc[ct][r] + pb2[row]);
      }
  }
}

// ---------------------------------------------------------------------------
// Kernel 5: u = bf16(q - gather(k) + pe), written in per-block (4-point)
// swizzled tile order: base = (b*512 + n>>2)*16384, tile (kc=c>>5)*4 + (n&3),
// element sw_b(c&31, kk).
// ---------------------------------------------------------------------------
__global__ __launch_bounds__(256) void build_u(const float* __restrict__ q,
                                               const float* __restrict__ kmat,
                                               const bf16* __restrict__ pe,
                                               const int* __restrict__ idx,
                                               bf16* __restrict__ u) {
  size_t gid = (size_t)blockIdx.x * 256 + threadIdx.x;   // (b,c,n)
  int b = (int)(gid >> 19);
  int rem = (int)(gid & ((1 << 19) - 1));
  int c = rem >> 11, n = rem & (NPTS - 1);
  float qv = q[gid];
  const float* kb = kmat + ((size_t)b * DIM + c) * NPTS;
  const int* ib = idx + ((size_t)b * NPTS + n) * KNN;
  bf16* ub = u + ((size_t)(b * 512 + (n >> 2)) * 16384)
               + ((c >> 5) * 4 + (n & 3)) * TILE;
#pragma unroll
  for (int kk = 0; kk < KNN; ++kk) {
    int j = ib[kk];
    ub[sw_b(c & 31, kk)] = (bf16)(qv - kb[j] + (float)pe[gid * KNN + kk]);
  }
}

// ---------------------------------------------------------------------------
// Kernel 6: fused attention MLP + softmax_one + aggregation.
// Block = 4 points (64 cols); hidden streamed in chunks of 64.
// Per-wave accumulators: 8 x v8f = 64 VGPRs (no spills).
// LDS: us 32K + W1 32K + H 8K + W2 32K = 104 KB (dynamic).
// ---------------------------------------------------------------------------
__global__ __launch_bounds__(256, 1) void attn_fused(
    const bf16* __restrict__ u,  const float* __restrict__ vx,
    const bf16* __restrict__ pe,
    const float* __restrict__ aW1, const float* __restrict__ ab1,
    const float* __restrict__ ag,  const float* __restrict__ abe,
    const float* __restrict__ am,  const float* __restrict__ av,
    const float* __restrict__ aW2, const float* __restrict__ ab2,
    float* __restrict__ agg) {
  extern __shared__ __attribute__((aligned(32))) char smem[];
  bf16* us  = (bf16*)smem;            // 32 tiles (kc*4+ct): 32 KB, LDS offset 0
  bf16* W1s = us  + 32 * TILE;        // 32 A-tiles (rt*8+kc): 32 KB
  bf16* Hsh = W1s + 32 * TILE;        //  8 B-tiles (kc2*4+ct): 8 KB
  bf16* W2s = Hsh + 8 * TILE;         // 32 A-tiles (rt*2+kc2): 32 KB

  int tid = threadIdx.x, wave = tid >> 5, lane = tid & 31;
  int b = blockIdx.x >> 9, ptile = blockIdx.x & 511;
  int p0 = ptile * 4;
  int mbase = (lane >> 4) << 3, nn = lane & 15;
  int wr0 = wave * 32;

  // ---- stage u tile: contiguous 32KB copy (pre-swizzled in HBM) ----
  const bf16* usrc = u + (size_t)(b * 512 + ptile) * 16384;
#if USE_ASYNC_LDS_COPY
#pragma unroll
  for (int i = 0; i < 8; ++i) {
    unsigned off = (unsigned)((i * 256 + tid) * 16);   // byte offset, b128 chunks
    unsigned lds_addr = off;                            // us sits at LDS offset 0
    unsigned long long ga = (unsigned long long)(const char*)usrc + off;
    asm volatile("global_load_async_to_lds_b128 %0, %1, off"
                 :: "v"(lds_addr), "v"(ga) : "memory");
  }
  asm volatile("s_wait_asynccnt 0" ::: "memory");
#else
#pragma unroll
  for (int i = 0; i < 8; ++i) {
    int o = (i * 256 + tid) * 8;
    *(v8bf*)(us + o) = *(const v8bf*)(usrc + o);
  }
#endif

  v8f Lacc[8];                        // wave owns 32 rows x 64 cols
#pragma unroll
  for (int i = 0; i < 8; ++i) Lacc[i] = v8f_zero();

  for (int hh = 0; hh < AH / 64; ++hh) {
    __syncthreads();
#pragma unroll
    for (int i = 0; i < 32; ++i) {   // aW1 chunk (64x256) as A-tile pairs
      int e2 = i * 256 + tid, r = e2 >> 7, c = (e2 & 127) << 1;
      const float* src = aW1 + (size_t)(hh * 64 + r) * DIM + c;
      v2bf p = {(bf16)src[0], (bf16)src[1]};
      *(v2bf*)(W1s + ((r >> 4) * 8 + (c >> 5)) * TILE + sw_a(r & 15, c & 31)) = p;
    }
#pragma unroll
    for (int i = 0; i < 32; ++i) {   // aW2 chunk (256x64) as A-tile pairs
      int e2 = i * 256 + tid, r = e2 >> 5, c = (e2 & 31) << 1;
      const float* src = aW2 + (size_t)r * AH + hh * 64 + c;
      v2bf p = {(bf16)src[0], (bf16)src[1]};
      *(v2bf*)(W2s + ((r >> 4) * 2 + (c >> 5)) * TILE + sw_a(r & 15, c & 31)) = p;
    }
    __syncthreads();

    // Phase A: H(64x64) = relu(bn(W1chunk @ u)); 16 tiles, 2 per wave
#pragma unroll
    for (int t2 = 0; t2 < 2; ++t2) {
      int tileid = wave * 2 + t2;
      int rt = tileid >> 2, ct = tileid & 3;
      v8f acc = v8f_zero();
#pragma unroll
      for (int kc = 0; kc < 8; ++kc)
        acc = wmma_bf16(frag_ld(W1s + (rt * 8 + kc) * TILE, lane),
                        frag_ld(us + (kc * 4 + ct) * TILE, lane), acc);
      v8bf hv;
#pragma unroll
      for (int r = 0; r < 8; ++r) {
        int ch = hh * 64 + rt * 16 + mbase + r;
        float sc = ag[ch] * rsqrtf(av[ch] + kEps);
        float xv = (acc[r] + ab1[ch] - am[ch]) * sc + abe[ch];
        hv[r] = (bf16)fmaxf(xv, 0.f);
      }
      // rows k = rt*16+mbase+r -> B-tile (kc2 = rt>>1), contiguous slots
      *(v8bf*)(Hsh + ((rt >> 1) * 4 + ct) * TILE + (nn + ((rt & 1) << 4)) * 16 + mbase) = hv;
    }
    __syncthreads();

    // Phase B: L += W2chunk(256x64) @ H(64x64); 2 rt x 4 ct tiles per wave
#pragma unroll
    for (int t = 0; t < 2; ++t) {
#pragma unroll
      for (int kc2 = 0; kc2 < 2; ++kc2) {
        v16bf a = frag_ld(W2s + ((wave * 2 + t) * 2 + kc2) * TILE, lane);
#pragma unroll
        for (int ct = 0; ct < 4; ++ct)
          Lacc[t * 4 + ct] =
              wmma_bf16(a, frag_ld(Hsh + (kc2 * 4 + ct) * TILE, lane), Lacc[t * 4 + ct]);
      }
    }
  }

  // Epilogue: softmax_one over K=16 (lanes of each half-wave), then
  // agg = sum_k attn * (v + pe).
#pragma unroll
  for (int t = 0; t < 2; ++t)
#pragma unroll
    for (int ct = 0; ct < 4; ++ct) {
      v8f Lv = Lacc[t * 4 + ct];
      int point = p0 + ct;
#pragma unroll
      for (int r = 0; r < 8; ++r) {
        int row = wr0 + t * 16 + mbase + r;
        float logit = Lv[r] + ab2[row];
        float m = logit;
#pragma unroll
        for (int off = 8; off >= 1; off >>= 1) m = fmaxf(m, __shfl_xor(m, off, 16));
        float e = __expf(logit - m);
        float s = e;
#pragma unroll
        for (int off = 8; off >= 1; off >>= 1) s += __shfl_xor(s, off, 16);
        float attn = e / (1.0f + s);
        size_t base = ((size_t)b * DIM + row) * NPTS + point;
        float val = vx[base] + (float)pe[base * KNN + nn];
        float prod = attn * val;
#pragma unroll
        for (int off = 8; off >= 1; off >>= 1) prod += __shfl_xor(prod, off, 16);
        if (nn == 0) agg[base] = prod;
      }
    }
}

// ---------------------------------------------------------------------------
// Kernel 7: y = We(128x256) @ agg + be + identity -> f32
// ---------------------------------------------------------------------------
__global__ __launch_bounds__(256, 1) void final_gemm(const float* __restrict__ agg,
                                                     const float* __restrict__ We,
                                                     const float* __restrict__ be,
                                                     const float* __restrict__ x,
                                                     float* __restrict__ out) {
  __shared__ __attribute__((aligned(32))) bf16 Wt[8 * TILE];
  __shared__ __attribute__((aligned(32))) bf16 At[4 * TILE];
  int tid = threadIdx.x, wave = tid >> 5, lane = tid & 31;
  int col0 = blockIdx.x * 64;
  int b = col0 >> 11, n0 = col0 & (NPTS - 1);

  v8f acc[4];
#pragma unroll
  for (int i = 0; i < 4; ++i) acc[i] = v8f_zero();

  for (int kc = 0; kc < DIM / 32; ++kc) {
    __syncthreads();
#pragma unroll
    for (int i = 0; i < 8; ++i) {
      int e2 = i * 256 + tid, r = e2 >> 4, c = (e2 & 15) << 1;
      const float* src = We + (size_t)r * DIM + kc * 32 + c;
      v2bf p = {(bf16)src[0], (bf16)src[1]};
      *(v2bf*)(Wt + (r >> 4) * TILE + sw_a(r & 15, c)) = p;
    }
#pragma unroll
    for (int i = 0; i < 8; ++i) {
      int e = i * 256 + tid, kr = e >> 6, nc = e & 63;
      At[(nc >> 4) * TILE + sw_b(kr, nc & 15)] =
          (bf16)agg[((size_t)b * DIM + kc * 32 + kr) * NPTS + n0 + nc];
    }
    __syncthreads();
    v16bf a = frag_ld(Wt + wave * TILE, lane);
#pragma unroll
    for (int ct = 0; ct < 4; ++ct)
      acc[ct] = wmma_bf16(a, frag_ld(At + ct * TILE, lane), acc[ct]);
  }
  int mbase = (lane >> 4) << 3, nn = lane & 15;
#pragma unroll
  for (int ct = 0; ct < 4; ++ct)
#pragma unroll
    for (int r = 0; r < 8; ++r) {
      int row = wave * 16 + mbase + r;
      int col = n0 + ct * 16 + nn;
      size_t o = ((size_t)b * CIN + row) * NPTS + col;
      out[o] = acc[ct][r] + be[row] + x[o];
    }
}

// ---------------------------------------------------------------------------
// Host launcher
// ---------------------------------------------------------------------------
extern "C" void kernel_launch(void* const* d_in, const int* in_sizes, int n_in,
                              void* d_out, int out_size, void* d_ws, size_t ws_size,
                              hipStream_t stream) {
  const float* x   = (const float*)d_in[0];
  const float* pos = (const float*)d_in[1];
  const float* Ws  = (const float*)d_in[2];
  const float* bs  = (const float*)d_in[3];
  const float* Wq  = (const float*)d_in[4];
  const float* bq  = (const float*)d_in[5];
  const float* Wk  = (const float*)d_in[6];
  const float* bk  = (const float*)d_in[7];
  const float* Wv  = (const float*)d_in[8];
  const float* bv  = (const float*)d_in[9];
  const float* pW1 = (const float*)d_in[10];
  const float* pb1 = (const float*)d_in[11];
  const float* pg  = (const float*)d_in[12];
  const float* pbe = (const float*)d_in[13];
  const float* pm  = (const float*)d_in[14];
  const float* pv  = (const float*)d_in[15];
  const float* pW2 = (const float*)d_in[16];
  const float* pb2 = (const float*)d_in[17];
  const float* aW1 = (const float*)d_in[18];
  const float* ab1 = (const float*)d_in[19];
  const float* ag  = (const float*)d_in[20];
  const float* abe = (const float*)d_in[21];
  const float* am  = (const float*)d_in[22];
  const float* av  = (const float*)d_in[23];
  const float* aW2 = (const float*)d_in[24];
  const float* ab2 = (const float*)d_in[25];
  const float* We  = (const float*)d_in[26];
  const float* be  = (const float*)d_in[27];

  char* ws = (char*)d_ws;
  size_t off = 0;
  int*   idx  = (int*)(ws + off);   off += (size_t)BN * KNN * 4;
  bf16*  h    = (bf16*)(ws + off);  off += (size_t)BN * DIM * 2;
  float* q    = (float*)(ws + off); off += (size_t)BN * DIM * 4;
  float* kmat = (float*)(ws + off); off += (size_t)BN * DIM * 4;
  float* vmat = (float*)(ws + off); off += (size_t)BN * DIM * 4;
  bf16*  pe   = (bf16*)(ws + off);  off += (size_t)BN * DIM * KNN * 2;
  bf16*  u    = (bf16*)(ws + off);  off += (size_t)BN * DIM * KNN * 2;
  float* agg  = (float*)(ws + off); off += (size_t)BN * DIM * 4;

  float* out = (float*)d_out;

  knn_topk<<<BN / 256, 256, 0, stream>>>(pos, idx);
  gemm_s<<<BN / 64, 256, 0, stream>>>(x, Ws, bs, h);
  gemm_qkv<<<BN / 64, 256, 0, stream>>>(h, Wq, bq, q);
  gemm_qkv<<<BN / 64, 256, 0, stream>>>(h, Wk, bk, kmat);
  gemm_qkv<<<BN / 64, 256, 0, stream>>>(h, Wv, bv, vmat);
  pe_mlp<<<BN / 8, 256, 0, stream>>>(pos, idx, pW1, pb1, pg, pbe, pm, pv, pW2, pb2, pe);
  build_u<<<(BN * DIM) / 256, 256, 0, stream>>>(q, kmat, pe, idx, u);

  size_t attn_lds = (size_t)(32 + 32 + 8 + 32) * TILE * sizeof(bf16);   // 104 KB
  attn_fused<<<BN / 4, 256, attn_lds, stream>>>(u, vmat, pe, aW1, ab1, ag, abe, am, av,
                                                aW2, ab2, agg);
  final_gemm<<<BN / 64, 256, 0, stream>>>(agg, We, be, x, out);
}
